// EMModel_85925115724026
// MI455X (gfx1250) — matrix operand order repaired and checked by reading, not compiled, not run on hardware
//
#include <hip/hip_runtime.h>
#include <hip/hip_bf16.h>

// ---------------------------------------------------------------------------
// EM recurrent model for MI455X (gfx1250, wave32, WMMA).
//   A=3 agents, EM=1024, ENTO=1024, B=64, T=128, 2 updates/step.
//   Per update/agent: GEMM M=64 x N=1024 x K<=4096, bf16 in / f32 acc, tanh.
//   Weights converted once to bf16 (24 MB, L2-resident), activations kept in
//   the concatenated-K layout so WMMA fragments load as contiguous b128s.
// ---------------------------------------------------------------------------

typedef __attribute__((ext_vector_type(16))) __bf16 v16bf;
typedef __attribute__((ext_vector_type(8)))  float  v8f;

#define A_N   3
#define EMD   1024
#define B_N   64
#define T_N   128
#define KTOT  4096   // x(1024) | vg_j0(1024) | vg_j1(1024) | v_self(1024)
#define ACTK  3072   // act buffer holds only the v parts (x read separately)

union Frag { uint4 q[2]; v16bf v; };

__device__ __forceinline__ uint4 ldg16(const __bf16* p) {
    return *reinterpret_cast<const uint4*>(p);
}

// scatter one tanh output v[b, a, o] into the 3 consumer slots of next act
__device__ __forceinline__ void scatter_act(__bf16* __restrict__ act,
                                            int a, int b, int o, __bf16 bv) {
    // self term: K-region 3 -> act offset 2048+o
    act[(size_t)(a * B_N + b) * ACTK + 2048 + o] = bv;
#pragma unroll
    for (int a2 = 0; a2 < A_N; ++a2) {
        if (a2 == a) continue;
        int j = a - (a > a2 ? 1 : 0);        // position of a in pre_idx[a2]
        act[(size_t)(a2 * B_N + b) * ACTK + j * EMD + o] = bv;
    }
}

// ---- one-time converters ---------------------------------------------------

// Wbf[a][o][k]: k<3072 -> W_ih[a][o][k] (= W_e | W_c), k>=3072 -> W_hh
__global__ void em_convert_w(const float* __restrict__ wih,
                             const float* __restrict__ whh,
                             __bf16* __restrict__ wbf) {
    int idx = blockIdx.x * blockDim.x + threadIdx.x;   // over 3*1024*4096
    int k  = idx & (KTOT - 1);
    int ao = idx >> 12;                                // a*1024 + o
    float v = (k < 3072) ? wih[(size_t)ao * 3072 + k]
                         : whh[(size_t)ao * EMD + (k - 3072)];
    wbf[idx] = (__bf16)v;
}

__global__ void em_convert_x(const float* __restrict__ x,
                             __bf16* __restrict__ xb) {
    int idx = blockIdx.x * blockDim.x + threadIdx.x;   // over T*B*ENTO
    xb[idx] = (__bf16)x[idx];
}

__global__ void em_init_act(const float* __restrict__ h0,
                            __bf16* __restrict__ act) {
    int idx = blockIdx.x * blockDim.x + threadIdx.x;   // over B*A*EM
    int b = idx / (A_N * EMD);
    int r = idx - b * (A_N * EMD);
    int a = r >> 10;
    int o = r & (EMD - 1);
    scatter_act(act, a, b, o, (__bf16)h0[idx]);
}

// ---- stride-specialized K loop --------------------------------------------
// abase: lane's A-row pointer (already includes k0, the (hi?8:0) sub-offset,
//        and the lane's row offset). wrow0/1 already include k0.
// Branch-free body: 12 x global_load_b128 + 8 x v_wmma per K-step of 32.

template <int ASTRIDE>
__device__ __forceinline__ void gemm_loop(const __bf16* __restrict__ abase,
                                          const __bf16* __restrict__ wrow0,
                                          const __bf16* __restrict__ wrow1,
                                          int ksteps, int hi, v8f acc[4][2]) {
#pragma unroll 2
    for (int s = 0; s < ksteps; ++s) {
        const int kb = s * 32 + (hi ? 8 : 0);    // B: 2 x 8 contiguous K / lane
        Frag bf[2];
        bf[0].q[0] = ldg16(wrow0 + kb);
        bf[0].q[1] = ldg16(wrow0 + kb + 16);
        bf[1].q[0] = ldg16(wrow1 + kb);
        bf[1].q[1] = ldg16(wrow1 + kb + 16);

        Frag af[4];
#pragma unroll
        for (int mi = 0; mi < 4; ++mi) {         // A: 2 x 8 contiguous K / lane
            const __bf16* p = abase + s * 32 + (size_t)mi * 16 * ASTRIDE;
            af[mi].q[0] = ldg16(p);
            af[mi].q[1] = ldg16(p + 16);
        }

#pragma unroll
        for (int mi = 0; mi < 4; ++mi)
#pragma unroll
            for (int ni = 0; ni < 2; ++ni)
                acc[mi][ni] = __builtin_amdgcn_wmma_f32_16x16x32_bf16(
                    false, af[mi].v, false, bf[ni].v,
                    (short)0, acc[mi][ni], false, false);
    }
}

// ---- the per-update GEMM+tanh kernel --------------------------------------
// grid = 96 blocks (a in [0,3) x ngroup in [0,32)), block = 128 (4 waves).
// Each wave: K-quarter, 4 m-tiles x 2 n-tiles of v_wmma_f32_16x16x32_bf16.
// LDS reduction across the 4 waves, then tanh + scatter.

__launch_bounds__(128)
__global__ void em_update(const __bf16* __restrict__ W,
                          const __bf16* __restrict__ actIn,
                          const __bf16* __restrict__ xIn,
                          __bf16* __restrict__ actOut,
                          float* __restrict__ outp,
                          int kStart) {
    __shared__ float red[4][32][8][8];   // [wave][lane][tile][r] = 32 KB

    const int lane = threadIdx.x & 31;
    const int wave = threadIdx.x >> 5;
    const int a    = blockIdx.x >> 5;          // 96 / 32
    const int ng   = blockIdx.x & 31;
    const int nbase = ng * 32;
    const int hi   = (lane >= 16) ? 1 : 0;
    const int l15  = lane & 15;

    const int kLen   = KTOT - kStart;
    const int chunk  = kLen >> 2;              // per-wave K span (1024 or 768)
    const int k0     = kStart + wave * chunk;
    const int ksteps = chunk >> 5;             // 32 K per wmma step

    v8f acc[4][2];
#pragma unroll
    for (int mi = 0; mi < 4; ++mi)
#pragma unroll
        for (int ni = 0; ni < 2; ++ni)
#pragma unroll
            for (int r = 0; r < 8; ++r) acc[mi][ni][r] = 0.0f;

    // B rows: lane holds output column n = nbase + ni*16 + l15, contiguous K
    const __bf16* wrow0 = W + ((size_t)(a * EMD + nbase + l15) * KTOT) + k0;
    const __bf16* wrow1 = wrow0 + (size_t)16 * KTOT;

    // The x-region [0,1024) is covered exactly by wave 0 when kStart==0;
    // every other (wave, kStart) combination lies wholly in the act region.
    if (k0 < EMD) {
        const __bf16* abase = xIn + (size_t)l15 * EMD + k0 + (hi ? 8 : 0);
        gemm_loop<EMD>(abase, wrow0, wrow1, ksteps, hi, acc);
    } else {
        const __bf16* abase =
            actIn + (size_t)(a * B_N + l15) * ACTK + (k0 - EMD) + (hi ? 8 : 0);
        gemm_loop<ACTK>(abase, wrow0, wrow1, ksteps, hi, acc);
    }

    // publish partials
#pragma unroll
    for (int t = 0; t < 8; ++t) {
        const int mi = t >> 1, ni = t & 1;
#pragma unroll
        for (int r = 0; r < 8; ++r) red[wave][lane][t][r] = acc[mi][ni][r];
    }
    __syncthreads();

    // wave w reduces tiles 2w and 2w+1, applies tanh, scatters
#pragma unroll
    for (int ti = 0; ti < 2; ++ti) {
        const int t  = wave * 2 + ti;
        const int mi = t >> 1, ni = t & 1;
        const int bb0 = mi * 16 + (hi ? 8 : 0);
        const int o   = nbase + ni * 16 + l15;
#pragma unroll
        for (int r = 0; r < 8; ++r) {
            float s = red[0][lane][t][r] + red[1][lane][t][r] +
                      red[2][lane][t][r] + red[3][lane][t][r];
            float v = tanhf(s);
            const int b = bb0 + r;
            scatter_act(actOut, a, b, o, (__bf16)v);
            if (outp) outp[(size_t)b * (A_N * EMD) + a * EMD + o] = v;
        }
    }
}

// ---------------------------------------------------------------------------

extern "C" void kernel_launch(void* const* d_in, const int* in_sizes, int n_in,
                              void* d_out, int out_size, void* d_ws, size_t ws_size,
                              hipStream_t stream) {
    (void)in_sizes; (void)n_in; (void)out_size; (void)ws_size;
    const float* ento = (const float*)d_in[0];   // (T, B, ENTO)
    const float* h0   = (const float*)d_in[1];   // (B, A, EM)
    const float* wih  = (const float*)d_in[2];   // (A, EM, 3072)
    const float* whh  = (const float*)d_in[3];   // (A, EM, EM)
    float* out = (float*)d_out;                  // (T, B, A*EM)

    char* ws = (char*)d_ws;
    __bf16* Wbf  = (__bf16*)ws;                              // 25,165,824 B
    __bf16* Xbf  = (__bf16*)(ws + (size_t)25165824);         // 16,777,216 B
    __bf16* ActA = (__bf16*)(ws + (size_t)41943040);         //  1,179,648 B
    __bf16* ActB = (__bf16*)(ws + (size_t)43122688);         //  1,179,648 B

    em_convert_w<<<(A_N * EMD * KTOT) / 256, 256, 0, stream>>>(wih, whh, Wbf);
    em_convert_x<<<(T_N * B_N * EMD) / 256, 256, 0, stream>>>(ento, Xbf);
    em_init_act<<<(B_N * A_N * EMD) / 256, 256, 0, stream>>>(h0, ActA);

    for (int t = 0; t < T_N; ++t) {
        // update 1: x = ento_feat[t], full K = 4096
        em_update<<<96, 128, 0, stream>>>(Wbf, ActA, Xbf + (size_t)t * B_N * EMD,
                                          ActB, nullptr, 0);
        // update 2: x = 0 -> skip x-region (K = 3072), write step output
        em_update<<<96, 128, 0, stream>>>(Wbf, ActB, Xbf,
                                          ActA, out + (size_t)t * B_N * A_N * EMD, 1024);
    }
}
